// MultiheadLatentAttn_54674933678912
// MI455X (gfx1250) — matrix operand order repaired and checked by reading, not compiled, not run on hardware
//
#include <hip/hip_runtime.h>
#include <hip/hip_bf16.h>

// ---------------------------------------------------------------------------
// MLA forward for MI455X (gfx1250): bf16 WMMA (f32 acc), absorbed latent form,
// flash softmax, LDS-staged C->A fragment re-layout.
// Round 2: kill scratch spills -> <=4 C-tiles per wave in GEMMs, norms split
// into separate kernels, __launch_bounds__(32,1) on all one-wave kernels.
// ---------------------------------------------------------------------------

typedef __bf16 bf16_t;
typedef __attribute__((ext_vector_type(16))) __bf16 v16bf;
typedef __attribute__((ext_vector_type(8)))  __bf16 v8bf;
typedef __attribute__((ext_vector_type(8)))  float  v8f;

#define NB   4
#define NS   2048
#define ND   1024
#define NH   16
#define DLQ  256
#define DLKV 128
#define DQK  20
#define DV   64
#define DCAT 160          // 128 latent + 10 rope + 22 zero pad
#define NKV  144          // wkv_a rows padded 138 -> 144
#define NTOK (NB*NS)      // 8192 flattened tokens

static __device__ __forceinline__ v8f wmma_bf16(v16bf a, v16bf b, v8f c) {
  return __builtin_amdgcn_wmma_f32_16x16x32_bf16(false, a, false, b,
                                                 (short)0, c, false, false);
}

static __device__ __forceinline__ v8f vzero8() {
  v8f z;
#pragma unroll
  for (int i = 0; i < 8; ++i) z[i] = 0.0f;
  return z;
}

// A fragment (16x32, row-major MxK, ld in elements; ld % 8 == 0).
// lane m=lane&15, half=lane>>4; VGPR v holds K=(v>>2)*16+half*8+(v&3)*2 (+1)
static __device__ __forceinline__ v16bf load_a16x32(const bf16_t* base, int ld) {
  const int lane = threadIdx.x & 31;
  const int m = lane & 15, h2 = lane >> 4;
  const bf16_t* r = base + m * ld + h2 * 8;
  v8bf lo = *(const v8bf*)(r);
  v8bf hi = *(const v8bf*)(r + 16);
  v16bf a;
#pragma unroll
  for (int i = 0; i < 8; ++i) { a[i] = lo[i]; a[8 + i] = hi[i]; }
  return a;
}

// B fragment (32x16 KxN) from N-major storage src[n][k] (ld % 16 == 0):
// two 16-byte loads per lane, K = half*8 + (0..7) and 16 + half*8 + (0..7).
static __device__ __forceinline__ v16bf load_bT16x32(const bf16_t* base, int ld) {
  const int lane = threadIdx.x & 31;
  const int n = lane & 15, h2 = lane >> 4;
  const bf16_t* r = base + n * ld + h2 * 8;
  v8bf lo = *(const v8bf*)(r);
  v8bf hi = *(const v8bf*)(r + 16);
  v16bf b;
#pragma unroll
  for (int i = 0; i < 8; ++i) { b[i] = lo[i]; b[8 + i] = hi[i]; }
  return b;
}

// reductions across the 16 lanes of a half-wave (rows live per C-elem v)
static __device__ __forceinline__ float hmax16(float x) {
#pragma unroll
  for (int d = 1; d < 16; d <<= 1) x = fmaxf(x, __shfl_xor(x, d, 32));
  return x;
}
static __device__ __forceinline__ float hsum16(float x) {
#pragma unroll
  for (int d = 1; d < 16; d <<= 1) x += __shfl_xor(x, d, 32);
  return x;
}

// ------------------------- prep / conversion kernels -----------------------

__global__ void k_cvt_bf16(const float* __restrict__ in, bf16_t* __restrict__ out, int n) {
  int i = blockIdx.x * blockDim.x + threadIdx.x;
  int st = gridDim.x * blockDim.x;
  for (; i < n; i += st) out[i] = (bf16_t)in[i];
}

// wkv_a (138x1024) -> bf16, zero-padded to 144 rows
__global__ void k_prep_wkva(const float* __restrict__ w, bf16_t* __restrict__ out) {
  int i = blockIdx.x * blockDim.x + threadIdx.x;
  int st = gridDim.x * blockDim.x;
  for (; i < NKV * ND; i += st) {
    int row = i >> 10;
    out[i] = (row < 138) ? (bf16_t)w[i] : (bf16_t)0.0f;
  }
}

// value weights: wv[h][v][c] = wkv_b[(h*74 + 10 + v)*128 + c]  (16,64,128) bf16
__global__ void k_prep_wv(const float* __restrict__ wkvb, bf16_t* __restrict__ out) {
  int i = blockIdx.x * blockDim.x + threadIdx.x;
  int st = gridDim.x * blockDim.x;
  for (; i < NH * DV * DLKV; i += st) {
    int c = i & 127;
    int v = (i >> 7) & 63;
    int h = i >> 13;
    out[i] = (bf16_t)wkvb[(h * 74 + 10 + v) * 128 + c];
  }
}

// ---------------- generic one-wave GEMM: C(16 x 4*16) = A @ W^T + bias -----
// A: bf16 (M x K) row-major, W: bf16 (N x K) row-major; out f32 (M x N)

template <int KDIM, int NDIM>
__global__ __launch_bounds__(32, 1)
void k_gemm_nt4(const bf16_t* __restrict__ Abuf, const bf16_t* __restrict__ Wbuf,
                const float* __restrict__ bias, int bias_n, float* __restrict__ out) {
  const int mt = blockIdx.x;              // 16-row tile
  const int nt0 = blockIdx.y * 4;         // 4 consecutive 16-col tiles
  const bf16_t* Ab = Abuf + (size_t)mt * 16 * KDIM;
  v8f acc[4];
#pragma unroll
  for (int t = 0; t < 4; ++t) acc[t] = vzero8();
  for (int k0 = 0; k0 < KDIM; k0 += 32) {
    v16bf a = load_a16x32(Ab + k0, KDIM);
#pragma unroll
    for (int t = 0; t < 4; ++t)
      acc[t] = wmma_bf16(a, load_bT16x32(Wbuf + (size_t)((nt0 + t) * 16) * KDIM + k0, KDIM),
                         acc[t]);
  }
  const int lane = threadIdx.x & 31, n = lane & 15, h2 = lane >> 4;
#pragma unroll
  for (int t = 0; t < 4; ++t)
#pragma unroll
    for (int v = 0; v < 8; ++v) {
      int col = (nt0 + t) * 16 + n;
      int row = mt * 16 + v + 8 * h2;
      float b = (col < bias_n) ? bias[col] : 0.0f;
      out[(size_t)row * NDIM + col] = acc[t][v] + b;
    }
}

// variant with 3 N-tiles (for N=144)
template <int KDIM, int NDIM>
__global__ __launch_bounds__(32, 1)
void k_gemm_nt3(const bf16_t* __restrict__ Abuf, const bf16_t* __restrict__ Wbuf,
                const float* __restrict__ bias, int bias_n, float* __restrict__ out) {
  const int mt = blockIdx.x;
  const int nt0 = blockIdx.y * 3;
  const bf16_t* Ab = Abuf + (size_t)mt * 16 * KDIM;
  v8f acc[3];
#pragma unroll
  for (int t = 0; t < 3; ++t) acc[t] = vzero8();
  for (int k0 = 0; k0 < KDIM; k0 += 32) {
    v16bf a = load_a16x32(Ab + k0, KDIM);
#pragma unroll
    for (int t = 0; t < 3; ++t)
      acc[t] = wmma_bf16(a, load_bT16x32(Wbuf + (size_t)((nt0 + t) * 16) * KDIM + k0, KDIM),
                         acc[t]);
  }
  const int lane = threadIdx.x & 31, n = lane & 15, h2 = lane >> 4;
#pragma unroll
  for (int t = 0; t < 3; ++t)
#pragma unroll
    for (int v = 0; v < 8; ++v) {
      int col = (nt0 + t) * 16 + n;
      int row = mt * 16 + v + 8 * h2;
      float b = (col < bias_n) ? bias[col] : 0.0f;
      out[(size_t)row * NDIM + col] = acc[t][v] + b;
    }
}

// -------- RMS norm over 256 cols -> bf16 (one 256-thread block per row) ----

__global__ void k_norm_q(const float* __restrict__ ql, const float* __restrict__ nw,
                         bf16_t* __restrict__ out) {
  const int row = blockIdx.x;
  const int c = threadIdx.x;               // 256
  float x = ql[(size_t)row * DLQ + c];
  float ss = x * x;
#pragma unroll
  for (int d = 1; d < 32; d <<= 1) ss += __shfl_xor(ss, d, 32);
  __shared__ float ps[8];
  if ((threadIdx.x & 31) == 0) ps[threadIdx.x >> 5] = ss;
  __syncthreads();
  float tot = 0.0f;
#pragma unroll
  for (int i = 0; i < 8; ++i) tot += ps[i];
  float r = rsqrtf(tot * (1.0f / DLQ) + 1e-6f);
  out[(size_t)row * DLQ + c] = (bf16_t)(x * r * nw[c]);
}

// -------- kv norm: rms over 128, emit kcat (160) + kcatT -------------------

__global__ void k_norm_kv(const float* __restrict__ kvl, const float* __restrict__ nw,
                          bf16_t* __restrict__ kcat, bf16_t* __restrict__ kcatT) {
  const int row = blockIdx.x;              // token
  const int c = threadIdx.x;               // 256 threads, 160 write
  float x = (c < NKV) ? kvl[(size_t)row * NKV + c] : 0.0f;
  float ss = (c < DLKV) ? x * x : 0.0f;
#pragma unroll
  for (int d = 1; d < 32; d <<= 1) ss += __shfl_xor(ss, d, 32);
  __shared__ float ps[8];
  if ((threadIdx.x & 31) == 0) ps[threadIdx.x >> 5] = ss;
  __syncthreads();
  float tot = 0.0f;
#pragma unroll
  for (int i = 0; i < 8; ++i) tot += ps[i];
  float r = rsqrtf(tot * (1.0f / DLKV) + 1e-6f);
  if (c < DCAT) {
    float val = (c < DLKV) ? x * r * nw[c] : ((c < 138) ? x : 0.0f);
    bf16_t bv = (bf16_t)val;
    int bb = row >> 11, s = row & 2047;
    kcat[(size_t)row * DCAT + c] = bv;
    kcatT[((size_t)bb * DCAT + c) * NS + s] = bv;
  }
}

// -------- absorb: q_cat[g,h,0:128] = scale * q_nope @ wkv_b_nope; rope tail -

__global__ void k_absorb(const float* __restrict__ q, const float* __restrict__ wkvb,
                         const float* __restrict__ ssmax, bf16_t* __restrict__ qcat) {
  const float alpha = rsqrtf((float)DQK) * ssmax[0] * logf((float)NS);
  int i = blockIdx.x * blockDim.x + threadIdx.x;
  int st = gridDim.x * blockDim.x;
  for (; i < NTOK * NH * DCAT; i += st) {
    int c = i % DCAT;
    int h = (i / DCAT) % NH;
    int g = i / (DCAT * NH);
    float val = 0.0f;
    if (c < DLKV) {
      const float* qrow = q + (size_t)g * (NH * DQK) + h * DQK;
      const float* w = wkvb + (size_t)(h * 74) * DLKV + c;
#pragma unroll
      for (int j = 0; j < 10; ++j) val += qrow[j] * w[j * DLKV];
    } else if (c < 138) {
      val = q[(size_t)g * (NH * DQK) + h * DQK + 10 + (c - DLKV)];
    }
    qcat[i] = (bf16_t)(val * alpha);
  }
}

// -------- flash attention + fused per-head value projection ----------------
// grid = (S/16, H, B), block = 32 (one wave per 16-query tile per head)

__global__ __launch_bounds__(32, 1)
void k_attn(const bf16_t* __restrict__ qcat, const bf16_t* __restrict__ kcat,
            const bf16_t* __restrict__ kcatT, const bf16_t* __restrict__ wv,
            bf16_t* __restrict__ vcat) {
  __shared__ __attribute__((aligned(32))) bf16_t Ptile[16 * 32];
  __shared__ __attribute__((aligned(32))) bf16_t Vlat[16 * DLKV];
  const int qt = blockIdx.x, h = blockIdx.y, b = blockIdx.z;
  const int s0 = qt * 16;
  const int lane = threadIdx.x & 31, n = lane & 15, h2 = lane >> 4;

  const bf16_t* Aq = qcat + ((size_t)(b * NS + s0) * NH + h) * DCAT;
  v16bf aq[5];
#pragma unroll
  for (int kk = 0; kk < 5; ++kk) aq[kk] = load_a16x32(Aq + kk * 32, NH * DCAT);

  v8f vacc[8];
#pragma unroll
  for (int t = 0; t < 8; ++t) vacc[t] = vzero8();
  float mrun[8], lrun[8];
#pragma unroll
  for (int v = 0; v < 8; ++v) { mrun[v] = -3.0e38f; lrun[v] = 0.0f; }

  const bf16_t* Kb  = kcat  + (size_t)b * NS * DCAT;
  const bf16_t* KTb = kcatT + (size_t)b * DCAT * NS;
  const int ntile = qt + 1;                         // causal: t-tiles 0..qt

  for (int p = 0; p < ntile; p += 2) {
    v8f st0 = vzero8(), st1 = vzero8();
#pragma unroll
    for (int kk = 0; kk < 5; ++kk)
      st0 = wmma_bf16(aq[kk], load_bT16x32(Kb + (size_t)(p * 16) * DCAT + kk * 32, DCAT), st0);
    const bool has1 = (p + 1) < ntile;              // uniform branch: EXEC stays full
    if (has1) {
#pragma unroll
      for (int kk = 0; kk < 5; ++kk)
        st1 = wmma_bf16(aq[kk], load_bT16x32(Kb + (size_t)((p + 1) * 16) * DCAT + kk * 32, DCAT), st1);
    }
    float fac[8];
#pragma unroll
    for (int v = 0; v < 8; ++v) {
      const int srow = s0 + v + 8 * h2;
      st0[v] = ((p * 16 + n) > srow) ? -3.0e38f : st0[v];
      st1[v] = (!has1 || ((p + 1) * 16 + n) > srow) ? -3.0e38f : st1[v];
      float mnew = fmaxf(mrun[v], hmax16(fmaxf(st0[v], st1[v])));
      float e0 = __expf(st0[v] - mnew);
      float e1 = __expf(st1[v] - mnew);
      fac[v] = __expf(mrun[v] - mnew);
      lrun[v] = lrun[v] * fac[v] + hsum16(e0 + e1);
      mrun[v] = mnew;
      const int pr = v + 8 * h2;
      Ptile[pr * 32 + n] = (bf16_t)e0;
      Ptile[pr * 32 + 16 + n] = (bf16_t)e1;
    }
#pragma unroll
    for (int ct = 0; ct < 8; ++ct)
#pragma unroll
      for (int v = 0; v < 8; ++v) vacc[ct][v] *= fac[v];
    __syncthreads();
    v16bf ap = load_a16x32(Ptile, 32);
#pragma unroll
    for (int ct = 0; ct < 8; ++ct)
      vacc[ct] = wmma_bf16(ap, load_bT16x32(KTb + (size_t)(ct * 16) * NS + p * 16, NS), vacc[ct]);
    __syncthreads();
  }

  // finalize softmax and stage v_lat (16x128) for the fused V projection
#pragma unroll
  for (int ct = 0; ct < 8; ++ct)
#pragma unroll
    for (int v = 0; v < 8; ++v)
      Vlat[(v + 8 * h2) * DLKV + ct * 16 + n] = (bf16_t)(vacc[ct][v] / lrun[v]);
  __syncthreads();

  v8f oacc[4];
#pragma unroll
  for (int t = 0; t < 4; ++t) oacc[t] = vzero8();
  const bf16_t* Wv = wv + (size_t)h * DV * DLKV;
  for (int k0 = 0; k0 < DLKV; k0 += 32) {
    v16bf av = load_a16x32(Vlat + k0, DLKV);
#pragma unroll
    for (int nt = 0; nt < 4; ++nt)
      oacc[nt] = wmma_bf16(av, load_bT16x32(Wv + (size_t)(nt * 16) * DLKV + k0, DLKV), oacc[nt]);
  }
#pragma unroll
  for (int nt = 0; nt < 4; ++nt)
#pragma unroll
    for (int v = 0; v < 8; ++v) {
      int row = b * NS + s0 + v + 8 * h2;
      int col = h * DV + nt * 16 + n;
      vcat[(size_t)row * (NH * DV) + col] = (bf16_t)oacc[nt][v];
    }
}

// -------- o = vcat @ wo^T + b  -> f32 out (4 N-tiles per wave) -------------

__global__ __launch_bounds__(32, 1)
void k_out(const bf16_t* __restrict__ vcat, const bf16_t* __restrict__ wob,
           const float* __restrict__ bias, float* __restrict__ out) {
  const int mt = blockIdx.x;               // 512
  const int nt0 = blockIdx.y * 4;          // 16 * 4 tiles = 1024 cols
  const bf16_t* Ab = vcat + (size_t)mt * 16 * ND;
  v8f acc[4];
#pragma unroll
  for (int t = 0; t < 4; ++t) acc[t] = vzero8();
  for (int k0 = 0; k0 < ND; k0 += 32) {
    v16bf a = load_a16x32(Ab + k0, ND);
#pragma unroll
    for (int t = 0; t < 4; ++t)
      acc[t] = wmma_bf16(a, load_bT16x32(wob + (size_t)((nt0 + t) * 16) * ND + k0, ND), acc[t]);
  }
  const int lane = threadIdx.x & 31, n = lane & 15, h2 = lane >> 4;
#pragma unroll
  for (int t = 0; t < 4; ++t)
#pragma unroll
    for (int v = 0; v < 8; ++v) {
      int col = (nt0 + t) * 16 + n;
      int row = mt * 16 + v + 8 * h2;
      out[(size_t)row * ND + col] = acc[t][v] + bias[col];
    }
}

// ---------------------------------------------------------------------------

extern "C" void kernel_launch(void* const* d_in, const int* in_sizes, int n_in,
                              void* d_out, int out_size, void* d_ws, size_t ws_size,
                              hipStream_t stream) {
  (void)in_sizes; (void)n_in; (void)out_size; (void)ws_size;
  const float* h        = (const float*)d_in[0];
  const float* wq_a_w   = (const float*)d_in[5];
  const float* wq_a_b   = (const float*)d_in[6];
  const float* q_norm_w = (const float*)d_in[7];
  const float* wq_b_w   = (const float*)d_in[8];
  const float* wq_b_b   = (const float*)d_in[9];
  const float* wkv_a_w  = (const float*)d_in[10];
  const float* wkv_a_b  = (const float*)d_in[11];
  const float* kv_norm_w= (const float*)d_in[12];
  const float* wkv_b_w  = (const float*)d_in[13];
  const float* wo_w     = (const float*)d_in[14];
  const float* wo_b     = (const float*)d_in[15];
  const float* ssmax    = (const float*)d_in[16];
  float* out = (float*)d_out;

  char* w = (char*)d_ws;
  size_t off = 0;
  auto alloc = [&](size_t bytes) -> void* {
    void* p = w + off;
    off += (bytes + 255) & ~(size_t)255;
    return p;
  };
  bf16_t* h_bf    = (bf16_t*)alloc((size_t)NTOK * ND * 2);
  bf16_t* wqa_bf  = (bf16_t*)alloc((size_t)DLQ * ND * 2);
  bf16_t* wqb_bf  = (bf16_t*)alloc((size_t)NH * DQK * DLQ * 2);
  bf16_t* wkva_bf = (bf16_t*)alloc((size_t)NKV * ND * 2);
  bf16_t* wv_bf   = (bf16_t*)alloc((size_t)NH * DV * DLKV * 2);
  bf16_t* wo_bf   = (bf16_t*)alloc((size_t)ND * ND * 2);
  float*  qlat_f  = (float*) alloc((size_t)NTOK * DLQ * 4);
  float*  kvlat_f = (float*) alloc((size_t)NTOK * NKV * 4);
  bf16_t* qln_bf  = (bf16_t*)alloc((size_t)NTOK * DLQ * 2);
  float*  q_f32   = (float*) alloc((size_t)NTOK * NH * DQK * 4);
  bf16_t* qcat_bf = (bf16_t*)alloc((size_t)NTOK * NH * DCAT * 2);
  bf16_t* kcat_bf = (bf16_t*)alloc((size_t)NTOK * DCAT * 2);
  bf16_t* kcatT_bf= (bf16_t*)alloc((size_t)NB * DCAT * NS * 2);
  bf16_t* vcat_bf = (bf16_t*)alloc((size_t)NTOK * NH * DV * 2);

  k_cvt_bf16 <<<4096, 256, 0, stream>>>(h, h_bf, NTOK * ND);
  k_cvt_bf16 <<<512,  256, 0, stream>>>(wq_a_w, wqa_bf, DLQ * ND);
  k_cvt_bf16 <<<160,  256, 0, stream>>>(wq_b_w, wqb_bf, NH * DQK * DLQ);
  k_prep_wkva<<<288,  256, 0, stream>>>(wkv_a_w, wkva_bf);
  k_prep_wv  <<<256,  256, 0, stream>>>(wkv_b_w, wv_bf);
  k_cvt_bf16 <<<2048, 256, 0, stream>>>(wo_w, wo_bf, ND * ND);

  // q latent GEMM (K=1024, N=256) + norm
  k_gemm_nt4<ND, DLQ><<<dim3(NTOK / 16, DLQ / 64), 32, 0, stream>>>(
      h_bf, wqa_bf, wq_a_b, DLQ, qlat_f);
  k_norm_q<<<NTOK, 256, 0, stream>>>(qlat_f, q_norm_w, qln_bf);

  // kv latent GEMM (K=1024, N=144) + norm -> kcat / kcatT
  k_gemm_nt3<ND, NKV><<<dim3(NTOK / 16, 3), 32, 0, stream>>>(
      h_bf, wkva_bf, wkv_a_b, 138, kvlat_f);
  k_norm_kv<<<NTOK, 256, 0, stream>>>(kvlat_f, kv_norm_w, kcat_bf, kcatT_bf);

  // q = q_lat_n @ wq_b^T + b  (K=256, N=320)
  k_gemm_nt4<DLQ, NH * DQK><<<dim3(NTOK / 16, 5), 32, 0, stream>>>(
      qln_bf, wqb_bf, wq_b_b, NH * DQK, q_f32);

  k_absorb<<<8192, 256, 0, stream>>>(q_f32, wkv_b_w, ssmax, qcat_bf);
  k_attn  <<<dim3(NS / 16, NH, NB), 32, 0, stream>>>(qcat_bf, kcat_bf, kcatT_bf,
                                                     wv_bf, vcat_bf);
  k_out   <<<dim3(NTOK / 16, ND / 64), 32, 0, stream>>>(vcat_bf, wo_bf, wo_b, out);
}
